// GroupGraph_11458972746222
// MI455X (gfx1250) — matrix operand order repaired and checked by reading, not compiled, not run on hardware
//
#include <hip/hip_runtime.h>

// ---------------------------------------------------------------------------
// Types for CDNA5 WMMA (wave32, 16x16x32 bf16 -> f32 accumulate)
// ---------------------------------------------------------------------------
typedef __attribute__((ext_vector_type(16))) __bf16       v16bf;
typedef __attribute__((ext_vector_type(8)))  float        v8f;
typedef __attribute__((ext_vector_type(4)))  unsigned int v4u;

union Frag {
    v16bf v;
    v4u   q[2];
};

__device__ __forceinline__ unsigned int bf16r(float f) {
    unsigned int u = __float_as_uint(f);
    return (u + 0x7FFFu + ((u >> 16) & 1u)) >> 16;   // round-to-nearest-even
}
__device__ __forceinline__ unsigned int pk2(float lo, float hi) {
    return bf16r(lo) | (bf16r(hi) << 16);
}
// monotonic float -> signed-int map (total order), for atomicMax-based segment_max
__device__ __forceinline__ int ordenc(float f) {
    int i = __float_as_int(f);
    return i >= 0 ? i : (i ^ 0x7FFFFFFF);
}
#define ORD_NEG_INF ((int)0x807FFFFFu)   // ordenc(-inf)

// ---------------------------------------------------------------------------
// Weight pre-pack: fp32 row-major W[K][Ncols] -> bf16 B-fragment stream.
// Layout consumed by the WMMA loops:
//   idx = ((kt*NT + nt)*32 + lane)*8 + v
//   n = nt*16 + (lane&15); k = kt*32 + (lane&16) + 2v; u32 = {W[k][n], W[k+1][n]}
// ---------------------------------------------------------------------------
__global__ void pack_w_kernel(const float* __restrict__ W, unsigned int* __restrict__ out,
                              int K, int Ncols) {
    int idx = blockIdx.x * blockDim.x + threadIdx.x;
    int total = (K * Ncols) >> 1;
    if (idx >= total) return;
    int NT   = Ncols >> 4;
    int v    = idx & 7;
    int lane = (idx >> 3) & 31;
    int t    = idx >> 8;
    int nt   = t % NT;
    int kt   = t / NT;
    int n = nt * 16 + (lane & 15);
    int k = kt * 32 + (lane & 16) + 2 * v;
    out[idx] = pk2(W[(size_t)k * Ncols + n], W[(size_t)(k + 1) * Ncols + n]);
}

// Fold BatchNorm (eval stats) into per-channel scale/shift.
__global__ void bn_prep_kernel(const float* __restrict__ gamma, const float* __restrict__ beta,
                               const float* __restrict__ mean, const float* __restrict__ var,
                               float* __restrict__ s, float* __restrict__ t, int n) {
    int i = blockIdx.x * blockDim.x + threadIdx.x;
    if (i >= n) return;
    float sc = gamma[i] * rsqrtf(var[i] + 1e-5f);
    s[i] = sc;
    t[i] = beta[i] - mean[i] * sc;
}

// ---------------------------------------------------------------------------
// Generic GEMM: C[M,Ncols] = A[M,K](f32) @ Bpack(bf16) (+bias)
// 1 wave per (m-tile, n-tile). A converted to bf16 in LDS once per block.
// ---------------------------------------------------------------------------
__global__ void gemm_wmma_kernel(const float* __restrict__ A, const unsigned int* __restrict__ Bpack,
                                 const float* __restrict__ bias, float* __restrict__ C,
                                 int M, int K, int Ncols) {
    extern __shared__ unsigned int Atile[];          // 16 * (K/2) u32
    const int mt = blockIdx.x, nt = blockIdx.y;
    const int lane = threadIdx.x;                    // 32 threads
    const int K2 = K >> 1;
    const int row0 = mt * 16;

    for (int idx = lane; idx < 16 * K2; idx += 32) {
        int r = idx / K2, p = idx - r * K2;
        const float* ap = A + (size_t)(row0 + r) * K + 2 * p;
        Atile[idx] = pk2(ap[0], ap[1]);
    }
    __syncthreads();

    v8f acc = {};
    const int NT = Ncols >> 4;
    const int r = lane & 15;
    const int half = (lane & 16) >> 1;               // 0 or 8 (u32 units)
    const int KT = K >> 5;
    for (int kt = 0; kt < KT; ++kt) {
        Frag a, b;
        int ab = r * K2 + kt * 16 + half;
        a.q[0] = *(const v4u*)&Atile[ab];
        a.q[1] = *(const v4u*)&Atile[ab + 4];
        size_t bb = ((size_t)(kt * NT + nt) * 32 + lane) * 8;
        b.q[0] = *(const v4u*)&Bpack[bb];
        b.q[1] = *(const v4u*)&Bpack[bb + 4];
        acc = __builtin_amdgcn_wmma_f32_16x16x32_bf16(false, a.v, false, b.v,
                                                      (short)0, acc, false, false);
    }

    const int col = nt * 16 + (lane & 15);
    const float bv = bias ? bias[col] : 0.0f;
    const int rbase = row0 + ((lane & 16) >> 1);     // +8 for upper half-lanes
#pragma unroll
    for (int i = 0; i < 8; ++i)
        C[(size_t)(rbase + i) * Ncols + col] = acc[i] + bv;
}

// x0: main-person rows (r % 4 == 0) take the Wm projection
__global__ void select_rows_kernel(const float* __restrict__ m, const float* __restrict__ c,
                                   float* __restrict__ x0, int total) {
    int i = blockIdx.x * blockDim.x + threadIdx.x;
    if (i >= total) return;
    int r = i >> 6;                                  // H = 64
    x0[i] = ((r & 3) == 0) ? m[i] : c[i];
}

// ---------------------------------------------------------------------------
// GCN aggregation (symmetric norm, self-loops)
// ---------------------------------------------------------------------------
__global__ void deg_init_kernel(float* deg, int n) {
    int i = blockIdx.x * blockDim.x + threadIdx.x;
    if (i < n) deg[i] = 1.0f;                        // self-loop
}
__global__ void deg_accum_kernel(const int* __restrict__ dst, float* deg, int e) {
    int i = blockIdx.x * blockDim.x + threadIdx.x;
    if (i < e) atomicAdd(&deg[dst[i]], 1.0f);
}
__global__ void dinv_kernel(const float* deg, float* dinv, int n) {
    int i = blockIdx.x * blockDim.x + threadIdx.x;
    if (i < n) dinv[i] = rsqrtf(deg[i]);
}
__global__ void gcn_self_kernel(const float* __restrict__ h, const float* __restrict__ dinv,
                                const float* __restrict__ bias, float* __restrict__ out,
                                int n, int C) {
    int i = blockIdx.x * blockDim.x + threadIdx.x;
    if (i >= n * C) return;
    int node = i / C, c = i - node * C;
    float di = dinv[node];
    out[i] = h[i] * di * di + bias[c];
}
__global__ void gcn_edge_kernel(const float* __restrict__ h, const float* __restrict__ dinv,
                                const int* __restrict__ src, const int* __restrict__ dst,
                                float* __restrict__ out, int e, int C) {
    long long i = (long long)blockIdx.x * blockDim.x + threadIdx.x;
    if (i >= (long long)e * C) return;
    int edge = (int)(i / C), c = (int)(i - (long long)edge * C);
    int s = src[edge], d = dst[edge];
    atomicAdd(&out[(size_t)d * C + c], h[(size_t)s * C + c] * dinv[s] * dinv[d]);
}

// ---------------------------------------------------------------------------
// EdgeConv: h_e = ReLU(BN([x_dst, x_src - x_dst])) @ W, segment_max over dst.
// 16 edges per block, 8 waves x one 16-col n-tile each (out width fixed 128).
// ---------------------------------------------------------------------------
__global__ void ord_init_kernel(int* ord, int total) {
    int i = blockIdx.x * blockDim.x + threadIdx.x;
    if (i < total) ord[i] = ORD_NEG_INF;
}
__global__ void ord_final_kernel(const int* __restrict__ ord, float* __restrict__ out, int total) {
    int i = blockIdx.x * blockDim.x + threadIdx.x;
    if (i >= total) return;
    int m = ord[i];
    int b = (m >= 0) ? m : (m ^ 0x7FFFFFFF);
    float f = __int_as_float(b);
    out[i] = ((b & 0x7F800000) == 0x7F800000) ? 0.0f : f;   // !isfinite -> 0
}

__global__ void edgeconv_wmma_kernel(const float* __restrict__ G, int C,
                                     const int* __restrict__ src, const int* __restrict__ dst,
                                     const float* __restrict__ bnS, const float* __restrict__ bnT,
                                     const unsigned int* __restrict__ Wpack,
                                     int* __restrict__ ordOut) {
    extern __shared__ unsigned int sh[];             // 16*C u32 A-tile + 32 ints
    unsigned int* Atile = sh;
    int* sIdx = (int*)(sh + 16 * C);

    const int tile = blockIdx.x;
    const int tid  = threadIdx.x;                    // 256
    const int wave = tid >> 5, lane = tid & 31;

    if (tid < 16) {
        sIdx[tid]      = src[tile * 16 + tid];
        sIdx[16 + tid] = dst[tile * 16 + tid];
    }
    __syncthreads();

    // Build bf16 A-tile [16 edges][2C] with BN+ReLU fused (C u32 per edge row).
    const int total = 16 * C;
    for (int idx = tid; idx < total; idx += 256) {
        int e = idx / C, p = idx - e * C;
        int ch0 = 2 * p;
        int s = sIdx[e], d = sIdx[16 + e];
        float f0, f1;
        if (ch0 < C) {                               // x_i half
            f0 = G[(size_t)d * C + ch0];
            f1 = G[(size_t)d * C + ch0 + 1];
        } else {                                     // (x_j - x_i) half
            int c0 = ch0 - C;
            f0 = G[(size_t)s * C + c0]     - G[(size_t)d * C + c0];
            f1 = G[(size_t)s * C + c0 + 1] - G[(size_t)d * C + c0 + 1];
        }
        f0 = fmaxf(f0 * bnS[ch0]     + bnT[ch0],     0.0f);
        f1 = fmaxf(f1 * bnS[ch0 + 1] + bnT[ch0 + 1], 0.0f);
        Atile[(size_t)e * C + p] = pk2(f0, f1);
    }
    __syncthreads();

    const int nt = wave;                             // 8 n-tiles = 128 cols
    const int K2 = C;                                // u32 per A row (2C bf16)
    const int r = lane & 15;
    const int half = (lane & 16) >> 1;
    const int KT = C >> 4;                           // (2C)/32
    v8f acc = {};
    for (int kt = 0; kt < KT; ++kt) {
        Frag a, b;
        int ab = r * K2 + kt * 16 + half;
        a.q[0] = *(const v4u*)&Atile[ab];
        a.q[1] = *(const v4u*)&Atile[ab + 4];
        size_t bb = ((size_t)(kt * 8 + nt) * 32 + lane) * 8;
        b.q[0] = *(const v4u*)&Wpack[bb];
        b.q[1] = *(const v4u*)&Wpack[bb + 4];
        acc = __builtin_amdgcn_wmma_f32_16x16x32_bf16(false, a.v, false, b.v,
                                                      (short)0, acc, false, false);
    }

    const int col = nt * 16 + (lane & 15);
    const int ehalf = (lane & 16) >> 1;              // +8 for upper lanes
#pragma unroll
    for (int i = 0; i < 8; ++i) {
        int e = i + ehalf;
        int d = sIdx[16 + e];
        atomicMax(&ordOut[(size_t)d * 128 + col], ordenc(acc[i]));
    }
}

// Final FC: y = h @ Wfc + bfc, Wfc is [128,1]
__global__ void fc_kernel(const float* __restrict__ h, const float* __restrict__ Wfc,
                          const float* __restrict__ bfc, float* __restrict__ out, int n) {
    int i = blockIdx.x * blockDim.x + threadIdx.x;
    if (i >= n) return;
    float s = bfc[0];
#pragma unroll 8
    for (int c = 0; c < 128; ++c) s += h[(size_t)i * 128 + c] * Wfc[c];
    out[i] = s;
}

// ---------------------------------------------------------------------------
// Host side
// ---------------------------------------------------------------------------
static inline int cdiv(long long a, int b) { return (int)((a + b - 1) / b); }

extern "C" void kernel_launch(void* const* d_in, const int* in_sizes, int n_in,
                              void* d_out, int out_size, void* d_ws, size_t ws_size,
                              hipStream_t stream) {
    const int IN = 512, H = 64;
    const int N = in_sizes[0] / IN;                  // 20000
    const int E = in_sizes[1] / 2;                   // 320000

    const float* x   = (const float*)d_in[0];
    const int*   ei  = (const int*)d_in[1];
    const int*   src = ei;
    const int*   dstp = ei + E;
    const float* Wm  = (const float*)d_in[2];  const float* bm  = (const float*)d_in[3];
    const float* Wc  = (const float*)d_in[4];  const float* bc  = (const float*)d_in[5];
    const float* Wg1 = (const float*)d_in[6];  const float* bg1 = (const float*)d_in[7];
    const float* Wg2 = (const float*)d_in[8];  const float* bg2 = (const float*)d_in[9];
    const float* Wg3 = (const float*)d_in[10]; const float* bg3 = (const float*)d_in[11];
    const float* e1g = (const float*)d_in[12], *e1b = (const float*)d_in[13],
               * e1m = (const float*)d_in[14], *e1v = (const float*)d_in[15],
               * e1W = (const float*)d_in[16];
    const float* e2g = (const float*)d_in[17], *e2b = (const float*)d_in[18],
               * e2m = (const float*)d_in[19], *e2v = (const float*)d_in[20],
               * e2W = (const float*)d_in[21];
    const float* e3g = (const float*)d_in[22], *e3b = (const float*)d_in[23],
               * e3m = (const float*)d_in[24], *e3v = (const float*)d_in[25],
               * e3W = (const float*)d_in[26];
    const float* Wfc = (const float*)d_in[27]; const float* bfc = (const float*)d_in[28];
    float* out = (float*)d_out;

    // Workspace carve-out (256B aligned slices)
    char* wsb = (char*)d_ws;
    size_t off = 0;
    auto take = [&](size_t bytes) -> void* {
        void* p = wsb + off;
        off = (off + bytes + 255) & ~(size_t)255;
        return p;
    };
    float* x0   = (float*)take((size_t)N * 64 * 4);
    float* bufA = (float*)take((size_t)N * 256 * 4);
    float* bufB = (float*)take((size_t)N * 256 * 4);
    float* deg  = (float*)take((size_t)N * 4);
    float* dinv = (float*)take((size_t)N * 4);
    int*   ord  = (int*)take((size_t)N * 128 * 4);
    unsigned int* WmP  = (unsigned int*)take((size_t)512 * 64 * 2);
    unsigned int* WcP  = (unsigned int*)take((size_t)512 * 64 * 2);
    unsigned int* Wg1P = (unsigned int*)take((size_t)64  * 128 * 2);
    unsigned int* Wg2P = (unsigned int*)take((size_t)128 * 128 * 2);
    unsigned int* Wg3P = (unsigned int*)take((size_t)128 * 256 * 2);
    unsigned int* e1P  = (unsigned int*)take((size_t)512 * 128 * 2);
    unsigned int* e2P  = (unsigned int*)take((size_t)256 * 128 * 2);
    unsigned int* e3P  = (unsigned int*)take((size_t)256 * 128 * 2);
    float* s1 = (float*)take(512 * 4); float* t1 = (float*)take(512 * 4);
    float* s2 = (float*)take(256 * 4); float* t2 = (float*)take(256 * 4);
    float* s3 = (float*)take(256 * 4); float* t3 = (float*)take(256 * 4);
    (void)ws_size; (void)n_in; (void)out_size;

    const int TB = 256;
    // --- weight packing + BN folding (cheap, every launch: deterministic) ---
    pack_w_kernel<<<cdiv(512LL*64/2,  TB), TB, 0, stream>>>(Wm,  WmP,  512, 64);
    pack_w_kernel<<<cdiv(512LL*64/2,  TB), TB, 0, stream>>>(Wc,  WcP,  512, 64);
    pack_w_kernel<<<cdiv(64LL*128/2,  TB), TB, 0, stream>>>(Wg1, Wg1P, 64,  128);
    pack_w_kernel<<<cdiv(128LL*128/2, TB), TB, 0, stream>>>(Wg2, Wg2P, 128, 128);
    pack_w_kernel<<<cdiv(128LL*256/2, TB), TB, 0, stream>>>(Wg3, Wg3P, 128, 256);
    pack_w_kernel<<<cdiv(512LL*128/2, TB), TB, 0, stream>>>(e1W, e1P,  512, 128);
    pack_w_kernel<<<cdiv(256LL*128/2, TB), TB, 0, stream>>>(e2W, e2P,  256, 128);
    pack_w_kernel<<<cdiv(256LL*128/2, TB), TB, 0, stream>>>(e3W, e3P,  256, 128);
    bn_prep_kernel<<<cdiv(512, TB), TB, 0, stream>>>(e1g, e1b, e1m, e1v, s1, t1, 512);
    bn_prep_kernel<<<cdiv(256, TB), TB, 0, stream>>>(e2g, e2b, e2m, e2v, s2, t2, 256);
    bn_prep_kernel<<<cdiv(256, TB), TB, 0, stream>>>(e3g, e3b, e3m, e3v, s3, t3, 256);

    // --- degrees / norm (self-loops included) ---
    deg_init_kernel<<<cdiv(N, TB), TB, 0, stream>>>(deg, N);
    deg_accum_kernel<<<cdiv(E, TB), TB, 0, stream>>>(dstp, deg, E);
    dinv_kernel<<<cdiv(N, TB), TB, 0, stream>>>(deg, dinv, N);

    const int MT = N / 16;                            // 1250
    // --- input projections (bufA/bufB as temporaries), then row-select ---
    gemm_wmma_kernel<<<dim3(MT, 64/16),  32, 16*(512/2)*4, stream>>>(x,  WmP, bm, bufA, N, 512, 64);
    gemm_wmma_kernel<<<dim3(MT, 64/16),  32, 16*(512/2)*4, stream>>>(x,  WcP, bc, bufB, N, 512, 64);
    select_rows_kernel<<<cdiv((long long)N*64, TB), TB, 0, stream>>>(bufA, bufB, x0, N*64);

    // --- GCN layer 1: x0[64] -> 128 ---
    gemm_wmma_kernel<<<dim3(MT, 128/16), 32, 16*(64/2)*4,  stream>>>(x0, Wg1P, nullptr, bufA, N, 64, 128);
    gcn_self_kernel<<<cdiv((long long)N*128, TB), TB, 0, stream>>>(bufA, dinv, bg1, bufB, N, 128);
    gcn_edge_kernel<<<cdiv((long long)E*128, TB), TB, 0, stream>>>(bufA, dinv, src, dstp, bufB, E, 128);
    // --- GCN layer 2: 128 -> 128 ---
    gemm_wmma_kernel<<<dim3(MT, 128/16), 32, 16*(128/2)*4, stream>>>(bufB, Wg2P, nullptr, bufA, N, 128, 128);
    gcn_self_kernel<<<cdiv((long long)N*128, TB), TB, 0, stream>>>(bufA, dinv, bg2, bufB, N, 128);
    gcn_edge_kernel<<<cdiv((long long)E*128, TB), TB, 0, stream>>>(bufA, dinv, src, dstp, bufB, E, 128);
    // --- GCN layer 3: 128 -> 256 ---
    gemm_wmma_kernel<<<dim3(MT, 256/16), 32, 16*(128/2)*4, stream>>>(bufB, Wg3P, nullptr, bufA, N, 128, 256);
    gcn_self_kernel<<<cdiv((long long)N*256, TB), TB, 0, stream>>>(bufA, dinv, bg3, bufB, N, 256);
    gcn_edge_kernel<<<cdiv((long long)E*256, TB), TB, 0, stream>>>(bufA, dinv, src, dstp, bufB, E, 256);

    const int ET = E / 16;                            // 20000 edge tiles
    // --- EdgeConv 1: C=256 (feat 512) -> 128 ---
    ord_init_kernel<<<cdiv((long long)N*128, TB), TB, 0, stream>>>(ord, N*128);
    edgeconv_wmma_kernel<<<ET, 256, 16*256*4 + 128, stream>>>(bufB, 256, src, dstp, s1, t1, e1P, ord);
    ord_final_kernel<<<cdiv((long long)N*128, TB), TB, 0, stream>>>(ord, bufA, N*128);
    // --- EdgeConv 2: C=128 -> 128 ---
    ord_init_kernel<<<cdiv((long long)N*128, TB), TB, 0, stream>>>(ord, N*128);
    edgeconv_wmma_kernel<<<ET, 256, 16*128*4 + 128, stream>>>(bufA, 128, src, dstp, s2, t2, e2P, ord);
    ord_final_kernel<<<cdiv((long long)N*128, TB), TB, 0, stream>>>(ord, bufB, N*128);
    // --- EdgeConv 3: C=128 -> 128 ---
    ord_init_kernel<<<cdiv((long long)N*128, TB), TB, 0, stream>>>(ord, N*128);
    edgeconv_wmma_kernel<<<ET, 256, 16*128*4 + 128, stream>>>(bufB, 128, src, dstp, s3, t3, e3P, ord);
    ord_final_kernel<<<cdiv((long long)N*128, TB), TB, 0, stream>>>(ord, bufA, N*128);

    // --- Final FC ---
    fc_kernel<<<cdiv(N, TB), TB, 0, stream>>>(bufA, Wfc, bfc, out, N);
}
